// PrototypeLayer_88441966559991
// MI455X (gfx1250) — matrix operand order repaired and checked by reading, not compiled, not run on hardware
//
#include <hip/hip_runtime.h>
#include <hip/hip_bf16.h>

// ---------------------------------------------------------------------------
// PrototypeLayer for MI455X (gfx1250, wave32, WMMA).
//
// Roofline: W is 16*1024*14336 fp32 = 940 MB -> spaces GEMV is the whole
// problem (0.5 FLOP/byte, ~40us at 23.3 TB/s). We run it as a WMMA
// f32_16x16x32_f16 GEMM with the means vector replicated into all 16 A rows
// (compute waste is free; W bytes are read exactly once). W is scaled by 256
// when converting to f16 (raw values ~2e-5 would be f16 subnormals), and the
// f32 accumulator is rescaled at the end. Everything else is tiny.
// ---------------------------------------------------------------------------

typedef __attribute__((ext_vector_type(16))) _Float16 v16h;
typedef __attribute__((ext_vector_type(8)))  float    v8f;

#define NB_CLASSES 16
#define EMBED_DIM  1024
#define SUM_K      14
#define E_DIM      (EMBED_DIM * SUM_K)   // 14336
#define BATCH      512
#define W_SCALE    256.0f
#define W_INVSCALE (1.0f / 256.0f)

// ---------------------------------------------------------------------------
// Kernel 1: per-class means. thread t = (c, d); uniform branch within a wave
// (1024 consecutive threads share c). x read 16x but it is L2-resident (2MB).
// ---------------------------------------------------------------------------
__global__ void k_means(const float* __restrict__ x, const int* __restrict__ y,
                        float* __restrict__ means) {
  int t = blockIdx.x * blockDim.x + threadIdx.x;   // 0 .. 16383
  int c = t >> 10;
  int d = t & (EMBED_DIM - 1);
  float acc = 0.0f;
  int   cnt = 0;
  for (int b = 0; b < BATCH; ++b) {
    int yb = y[b];
    if (yb == c) {
      cnt += 1;
      acc += x[(size_t)b * EMBED_DIM + d];
    }
  }
  means[t] = acc / fmaxf((float)cnt, 1.0f);
}

// ---------------------------------------------------------------------------
// Kernel 2: spaces[c, e] = sum_d means[c,d] * W[c,d,e] + bias[c,e]
// One wave per 16-column tile over the full K=1024. 16*896 = 14336 waves.
// WMMA A = means replicated in all 16 rows; B = 32x16 tile of W (f16, x256).
// ---------------------------------------------------------------------------
__global__ void k_spaces_wmma(const float* __restrict__ W,
                              const float* __restrict__ means,
                              const float* __restrict__ bias,
                              float* __restrict__ spaces) {
  const int lane = threadIdx.x & 31;
  const int gwid = blockIdx.x * (blockDim.x >> 5) + (threadIdx.x >> 5);
  const int c    = gwid / (E_DIM / 16);          // class
  const int tile = gwid % (E_DIM / 16);          // 16-col tile within class
  const int e0   = tile * 16;
  const int n    = lane & 15;                    // WMMA column for this lane
  const int grp  = (lane >> 4) & 1;              // lane group 0/1

  const float* __restrict__ Wc = W     + (size_t)c * EMBED_DIM * E_DIM;
  const float* __restrict__ Mc = means + (size_t)c * EMBED_DIM;

  v8f acc = {};

  for (int k0 = 0; k0 < EMBED_DIM; k0 += 32) {
    // ---- A operand: contiguous K order matching B.
    // lanes 0-15: halfs 0..15 -> K=0..15
    // lanes16-31: halfs 0..15 -> K=16..31
    const float* Ma = Mc + k0 + grp * 16;
    float4 a0 = *(const float4*)(Ma + 0);
    float4 a1 = *(const float4*)(Ma + 4);
    float4 a2 = *(const float4*)(Ma + 8);
    float4 a3 = *(const float4*)(Ma + 12);
    v16h Av;
    Av[0]  = (_Float16)a0.x; Av[1]  = (_Float16)a0.y;
    Av[2]  = (_Float16)a0.z; Av[3]  = (_Float16)a0.w;
    Av[4]  = (_Float16)a1.x; Av[5]  = (_Float16)a1.y;
    Av[6]  = (_Float16)a1.z; Av[7]  = (_Float16)a1.w;
    Av[8]  = (_Float16)a2.x; Av[9]  = (_Float16)a2.y;
    Av[10] = (_Float16)a2.z; Av[11] = (_Float16)a2.w;
    Av[12] = (_Float16)a3.x; Av[13] = (_Float16)a3.y;
    Av[14] = (_Float16)a3.z; Av[15] = (_Float16)a3.w;

    // ---- B operand: 32x16, lanes 0-15 hold K=0..15 of col n, lanes 16-31
    // hold K=16..31 (contiguous K order, same as A). Each load instr: 2x64B row segments;
    // adjacent waves consume the other halves of the 128B lines.
    const float* Wb = Wc + (size_t)(k0 + grp * 16) * E_DIM + e0 + n;
    v16h Bv;
#pragma unroll
    for (int h = 0; h < 16; ++h)
      Bv[h] = (_Float16)(Wb[(size_t)h * E_DIM] * W_SCALE);

    // Prefetch next K-block's rows to keep HBM pipe full.
    if (k0 + 32 < EMBED_DIM)
      __builtin_prefetch(Wb + (size_t)32 * E_DIM, 0, 1);

    acc = __builtin_amdgcn_wmma_f32_16x16x32_f16(
        /*neg_a=*/false, Av, /*neg_b=*/false, Bv,
        /*c_mod=*/(short)0, acc, /*reuse_a=*/false, /*reuse_b=*/false);
  }

  // All 16 D rows are identical; lane l holds column (l&15) in acc[0].
  if (lane < 16) {
    size_t o = (size_t)c * E_DIM + e0 + n;
    spaces[o] = acc[0] * W_INVSCALE + bias[o];
  }
}

// ---------------------------------------------------------------------------
// Kernel 3: inverse L2 norms. One wave per row: rows 0..511 = x rows,
// rows 512..735 = the 224 prototype rows of spaces.
// ---------------------------------------------------------------------------
__global__ void k_norms(const float* __restrict__ x,
                        const float* __restrict__ spaces,
                        float* __restrict__ invx, float* __restrict__ invp) {
  int wid  = blockIdx.x * (blockDim.x >> 5) + (threadIdx.x >> 5);
  int lane = threadIdx.x & 31;
  const float* row;
  if (wid < BATCH) {
    row = x + (size_t)wid * EMBED_DIM;
  } else if (wid < BATCH + NB_CLASSES * SUM_K) {
    int r = wid - BATCH;                 // c*14 + k
    int c = r / SUM_K, k = r % SUM_K;
    row = spaces + (size_t)c * E_DIM + (size_t)k * EMBED_DIM;
  } else {
    return;
  }
  float s = 0.0f;
  for (int d = lane; d < EMBED_DIM; d += 32) {
    float v = row[d];
    s += v * v;
  }
  for (int off = 16; off >= 1; off >>= 1) s += __shfl_xor(s, off, 32);
  if (lane == 0) {
    if (wid < BATCH) invx[wid] = rsqrtf(s);
    else             invp[wid - BATCH] = rsqrtf(s);
  }
}

// ---------------------------------------------------------------------------
// Kernel 4a: cosine sims [512,14] + per-group first-max argmax. One wave per b.
// ---------------------------------------------------------------------------
__global__ void k_sims_argmax(const float* __restrict__ x,
                              const int* __restrict__ y,
                              const float* __restrict__ spaces,
                              const float* __restrict__ invx,
                              const float* __restrict__ invp,
                              int* __restrict__ selidx,
                              float* __restrict__ out_idx) {
  int b    = blockIdx.x * (blockDim.x >> 5) + (threadIdx.x >> 5);
  int lane = threadIdx.x & 31;
  if (b >= BATCH) return;
  int yb = y[b];
  const float* xr = x + (size_t)b * EMBED_DIM;
  const float* sp = spaces + (size_t)yb * E_DIM;

  float sims[SUM_K];
#pragma unroll
  for (int k = 0; k < SUM_K; ++k) {
    const float* pr = sp + (size_t)k * EMBED_DIM;
    float s = 0.0f;
    for (int d = lane; d < EMBED_DIM; d += 32) s += xr[d] * pr[d];
    for (int off = 16; off >= 1; off >>= 1) s += __shfl_xor(s, off, 32);
    sims[k] = s * invx[b] * invp[yb * SUM_K + k];
  }
  if (lane == 0) {
    const int offs[5] = {0, 2, 5, 9, 14};     // K_RANGE prefix sums
#pragma unroll
    for (int g = 0; g < 4; ++g) {
      int   best = offs[g];
      float bs   = sims[offs[g]];
      for (int k = offs[g] + 1; k < offs[g + 1]; ++k)
        if (sims[k] > bs) { bs = sims[k]; best = k; }   // first-max tiebreak
      selidx[b * 4 + g]  = best;
      out_idx[b * 4 + g] = (float)best;
    }
  }
}

// ---------------------------------------------------------------------------
// Kernel 4b: write prototype_set [512,14,1024] and nearest (normalized
// selected prototypes) [512,4,1024]. One block per batch row; spaces (0.9 MB)
// is L2-resident so the gathers are cheap; ~37 MB of streaming writes.
// ---------------------------------------------------------------------------
__global__ void k_gather(const int* __restrict__ y,
                         const float* __restrict__ spaces,
                         const float* __restrict__ invp,
                         const int* __restrict__ selidx,
                         float* __restrict__ out_near,
                         float* __restrict__ out_proto) {
  int b   = blockIdx.x;
  int tid = threadIdx.x;
  int yb  = y[b];
  const float* sp = spaces + (size_t)yb * E_DIM;
#pragma unroll
  for (int k = 0; k < SUM_K; ++k)
    for (int d = tid; d < EMBED_DIM; d += 256)
      out_proto[((size_t)b * SUM_K + k) * EMBED_DIM + d] =
          sp[(size_t)k * EMBED_DIM + d];
#pragma unroll
  for (int g = 0; g < 4; ++g) {
    int   sel = selidx[b * 4 + g];
    float s   = invp[yb * SUM_K + sel];
    for (int d = tid; d < EMBED_DIM; d += 256)
      out_near[((size_t)b * 4 + g) * EMBED_DIM + d] =
          sp[(size_t)sel * EMBED_DIM + d] * s;
  }
}

// ---------------------------------------------------------------------------
extern "C" void kernel_launch(void* const* d_in, const int* in_sizes, int n_in,
                              void* d_out, int out_size, void* d_ws, size_t ws_size,
                              hipStream_t stream) {
  const float* x    = (const float*)d_in[0];   // [512,1024]
  const int*   y    = (const int*)d_in[1];     // [512]
  const float* W    = (const float*)d_in[2];   // [16,1024,14336]
  const float* bias = (const float*)d_in[3];   // [16,14336]

  // Workspace layout (float units).
  float* ws      = (float*)d_ws;
  float* means   = ws;                                   // 16*1024
  float* spaces  = ws + NB_CLASSES * EMBED_DIM;          // 16*14336
  float* invx    = spaces + (size_t)NB_CLASSES * E_DIM;  // 512
  float* invp    = invx + BATCH;                         // 224
  int*   selidx  = (int*)(invp + NB_CLASSES * SUM_K);    // 512*4 ints

  // Output layout: nearest | prototype_set | indices (flat, return order).
  float* out_near  = (float*)d_out;                              // 512*4*1024
  float* out_proto = out_near + (size_t)BATCH * 4 * EMBED_DIM;   // 512*14*1024
  float* out_idx   = out_proto + (size_t)BATCH * SUM_K * EMBED_DIM; // 512*4

  // 1) per-class means: 16384 threads
  k_means<<<64, 256, 0, stream>>>(x, y, means);

  // 2) spaces via WMMA: 16*896 = 14336 waves, 8 waves/block -> 1792 blocks
  k_spaces_wmma<<<1792, 256, 0, stream>>>(W, means, bias, spaces);

  // 3) norms: 736 waves -> 92 blocks of 8 waves
  k_norms<<<92, 256, 0, stream>>>(x, spaces, invx, invp);

  // 4a) sims + argmax: 512 waves -> 64 blocks of 8 waves
  k_sims_argmax<<<64, 256, 0, stream>>>(x, y, spaces, invx, invp, selidx, out_idx);

  // 4b) gather/normalize outputs: one block per batch row
  k_gather<<<BATCH, 256, 0, stream>>>(y, spaces, invp, selidx, out_near, out_proto);
}
